// InvariantPointAttention_4870492913926
// MI455X (gfx1250) — matrix operand order repaired and checked by reading, not compiled, not run on hardware
//
#include <hip/hip_runtime.h>
#include <hip/hip_bf16.h>
#include <math.h>

typedef __attribute__((ext_vector_type(16))) _Float16 v16h;
typedef __attribute__((ext_vector_type(8)))  _Float16 v8h;
typedef __attribute__((ext_vector_type(8)))  float    v8f;

// ---- problem constants ----
constexpr int NH  = 8;     // heads
constexpr int SK  = 16;    // scalar qk dim
constexpr int SV  = 16;    // scalar v dim
constexpr int PK  = 4;     // point qk count
constexpr int PV  = 8;     // point v count
constexpr int NN  = 512;   // sequence length
constexpr int DIM = 384;
constexpr int PDIM = 128;
constexpr int RESW = NH * (SV + PDIM + PV * 4);  // 1408
constexpr float SCALE_SCALAR = 0.14433756729740643f;   // (3*16)^-0.5
constexpr float SCALE_POINT  = 0.13608276348795434f;   // (3*4*4.5)^-0.5
constexpr float SCALE_PAIR   = 0.5773502691896258f;    // 3^-0.5
constexpr float EPSN = 1e-8f;

// ---- workspace layout (float offsets) ----
constexpr int OFF_QS      = 0;                          // 512*128
constexpr int OFF_KS      = OFF_QS + NN * 128;
constexpr int OFF_VS      = OFF_KS + NN * 128;
constexpr int OFF_QP      = OFF_VS + NN * 128;          // 512*96  (local->global in place)
constexpr int OFF_KP      = OFF_QP + NN * 96;
constexpr int OFF_VP      = OFF_KP + NN * 96;           // 512*192
constexpr int OFF_LOGITS  = OFF_VP + NN * 192;          // 8*512*512
constexpr int OFF_RESULTS = OFF_LOGITS + NH * NN * NN;  // 512*1408
constexpr int OFF_RESPG   = OFF_RESULTS + NN * RESW;    // 512*192 (global-frame attended pts)
constexpr int OFF_RQ      = OFF_RESPG + NN * 192;       // 8*512
constexpr int OFF_CK      = OFF_RQ + NH * NN;           // 8*512
constexpr int OFF_F32_END = OFF_CK + NH * NN;
// f16 region follows (16-byte aligned: OFF_F32_END*4 is a multiple of 16)
constexpr int HQT = NH * NN * 32;  // Qtilde halves

__device__ __forceinline__ v8f wmma16(v16h a, v16h b, v8f c) {
  // D = A(16x32 f16) * B(32x16 f16) + C(16x16 f32)
  return __builtin_amdgcn_wmma_f32_16x16x32_f16(false, a, false, b, (short)0, c,
                                                false, false);
}

// Load 16-wide f16 A/B fragment from a contiguous 32-float f32 window:
// elems 0..15 = row[kh*16 .. +15]
__device__ __forceinline__ v16h load_frag_f32c(const float* __restrict__ row, int kh) {
  const float4* p0 = (const float4*)(row + kh * 16);
  const float4* p1 = (const float4*)(row + kh * 16 + 8);
  float4 u = p0[0], v = p0[1], w = p1[0], x = p1[1];
  v16h a;
  a[0]  = (_Float16)u.x; a[1]  = (_Float16)u.y; a[2]  = (_Float16)u.z; a[3]  = (_Float16)u.w;
  a[4]  = (_Float16)v.x; a[5]  = (_Float16)v.y; a[6]  = (_Float16)v.z; a[7]  = (_Float16)v.w;
  a[8]  = (_Float16)w.x; a[9]  = (_Float16)w.y; a[10] = (_Float16)w.z; a[11] = (_Float16)w.w;
  a[12] = (_Float16)x.x; a[13] = (_Float16)x.y; a[14] = (_Float16)x.z; a[15] = (_Float16)x.w;
  return a;
}

__device__ __forceinline__ v16h cat8(v8h lo, v8h hi) {
  return __builtin_shufflevector(lo, hi, 0, 1, 2, 3, 4, 5, 6, 7,
                                 8, 9, 10, 11, 12, 13, 14, 15);
}

// ======================================================================
// K1: generic X(512 x Kd, f32) @ W(Kd x Nd, f32) -> out (512 x Nd, f32)
// one wave per 16x16 output tile
// ======================================================================
__global__ void k_gemm_xw(const float* __restrict__ X, const float* __restrict__ W,
                          float* __restrict__ out, int Kd, int Nd) {
  const int lane = threadIdx.x;
  const int m0 = blockIdx.x * 16, n0 = blockIdx.y * 16;
  const int m = lane & 15, kh = lane >> 4, nn = lane & 15;
  v8f acc = {};
  for (int k0 = 0; k0 < Kd; k0 += 32) {
    v16h a = load_frag_f32c(X + (m0 + m) * Kd + k0, kh);
    v16h b;
#pragma unroll
    for (int e = 0; e < 16; ++e)
      b[e] = (_Float16)W[(k0 + kh * 16 + e) * Nd + n0 + nn];
    acc = wmma16(a, b, acc);
  }
#pragma unroll
  for (int r = 0; r < 8; ++r)
    out[(m0 + r + 8 * kh) * Nd + n0 + nn] = acc[r];
}

// ======================================================================
// K2: rotate points to global frame (in place): g = p . R + t
// ======================================================================
__global__ void k_rotate(float* __restrict__ p, const float* __restrict__ rot,
                         const float* __restrict__ trans, int ppr) {
  int t = blockIdx.x * blockDim.x + threadIdx.x;
  if (t >= NN * ppr) return;
  int i = t / ppr;
  const float* R = rot + i * 9;
  const float* T = trans + i * 3;
  float x = p[t * 3 + 0], y = p[t * 3 + 1], z = p[t * 3 + 2];
  p[t * 3 + 0] = x * R[0] + y * R[3] + z * R[6] + T[0];
  p[t * 3 + 1] = x * R[1] + y * R[4] + z * R[7] + T[1];
  p[t * 3 + 2] = x * R[2] + y * R[5] + z * R[8] + T[2];
}

// ======================================================================
// K3: build augmented f16 Qtilde/Ktilde (K=32) + row/col distance biases
// ======================================================================
__global__ void k_prep(const float* __restrict__ qs, const float* __restrict__ ks,
                       const float* __restrict__ qp, const float* __restrict__ kp,
                       const float* __restrict__ pwraw,
                       _Float16* __restrict__ Qt, _Float16* __restrict__ Kt,
                       float* __restrict__ rq, float* __restrict__ ck) {
  int t = blockIdx.x * blockDim.x + threadIdx.x;
  if (t >= NN * NH) return;
  int i = t / NH, h = t % NH;
  float c = log1pf(expf(pwraw[h])) * SCALE_POINT;  // softplus * scale
  _Float16* qr = Qt + (h * NN + i) * 32;
  _Float16* kr = Kt + (h * NN + i) * 32;
#pragma unroll
  for (int d = 0; d < 16; ++d) {
    qr[d] = (_Float16)(qs[i * 128 + h * 16 + d] * SCALE_SCALAR);
    kr[d] = (_Float16)ks[i * 128 + h * 16 + d];
  }
  float sq = 0.f, sk = 0.f;
#pragma unroll
  for (int d = 0; d < 12; ++d) {
    float qv = qp[i * 96 + h * 12 + d];
    float kv = kp[i * 96 + h * 12 + d];
    sq += qv * qv; sk += kv * kv;
    qr[16 + d] = (_Float16)(c * qv);
    kr[16 + d] = (_Float16)kv;
  }
#pragma unroll
  for (int d = 28; d < 32; ++d) { qr[d] = (_Float16)0.f; kr[d] = (_Float16)0.f; }
  rq[h * NN + i] = -0.5f * c * sq;
  ck[h * NN + i] = -0.5f * c * sk;
}

// ======================================================================
// K4: pair bias: logits[h][i*NN+j] = (pair[i,j,:] . Wpair[:,h] + bpair) * SCALE
//     GEMM (262144 x 128) @ (128 x 8-pad-16), WMMA, one wave / 16 rows
// ======================================================================
__global__ void k_pairbias(const float* __restrict__ pair, const float* __restrict__ Wp,
                           const float* __restrict__ bp, float* __restrict__ logits) {
  const int lane = threadIdx.x;
  const int m0 = blockIdx.x * 16;
  const int m = lane & 15, kh = lane >> 4, nn = lane & 15;
  v8f acc = {};
#pragma unroll
  for (int k0 = 0; k0 < PDIM; k0 += 32) {
    v16h a = load_frag_f32c(pair + (m0 + m) * PDIM + k0, kh);
    v16h b;
#pragma unroll
    for (int e = 0; e < 16; ++e)
      b[e] = (nn < NH) ? (_Float16)Wp[(k0 + kh * 16 + e) * NH + nn] : (_Float16)0.f;
    acc = wmma16(a, b, acc);
  }
  if (nn < NH) {
    float bb = bp[nn];
#pragma unroll
    for (int r = 0; r < 8; ++r)
      logits[nn * (NN * NN) + m0 + r + 8 * kh] = (acc[r] + bb) * SCALE_PAIR;
  }
}

// ======================================================================
// K5: scalar + point logits in one K=32 WMMA, accumulating onto pair bias
// ======================================================================
__global__ void k_qk_logits(const _Float16* __restrict__ Qt, const _Float16* __restrict__ Kt,
                            const float* __restrict__ rq, const float* __restrict__ ck,
                            float* __restrict__ logits) {
  const int lane = threadIdx.x;
  const int i0 = blockIdx.x * 16, j0 = blockIdx.y * 16, h = blockIdx.z;
  const int m = lane & 15, kh = lane >> 4, nn = lane & 15;
  const _Float16* qr = Qt + (h * NN + i0 + m) * 32;
  v16h a = cat8(*(const v8h*)(qr + kh * 16), *(const v8h*)(qr + kh * 16 + 8));
  const _Float16* kr = Kt + (h * NN + j0 + nn) * 32 + kh * 16;
  v16h b = cat8(*(const v8h*)(kr), *(const v8h*)(kr + 8));
  float* L = logits + h * (NN * NN);
  v8f cfr;
#pragma unroll
  for (int r = 0; r < 8; ++r) cfr[r] = L[(i0 + r + 8 * kh) * NN + j0 + nn];
  v8f d = wmma16(a, b, cfr);
  const float ckv = ck[h * NN + j0 + nn];
#pragma unroll
  for (int r = 0; r < 8; ++r) {
    int row = i0 + r + 8 * kh;
    L[row * NN + j0 + nn] = d[r] + rq[h * NN + row] + ckv;
  }
}

// ======================================================================
// K6: row softmax over j (row = h*NN + i), 128 threads x 4 elems
// ======================================================================
__global__ void k_softmax(float* __restrict__ logits) {
  float* L = logits + blockIdx.x * NN;
  const int t = threadIdx.x;
  __shared__ float red[128];
  float v[4];
  float mx = -1e30f;
#pragma unroll
  for (int c = 0; c < 4; ++c) { v[c] = L[t + c * 128]; mx = fmaxf(mx, v[c]); }
  red[t] = mx; __syncthreads();
  for (int s = 64; s > 0; s >>= 1) { if (t < s) red[t] = fmaxf(red[t], red[t + s]); __syncthreads(); }
  mx = red[0]; __syncthreads();
  float sm = 0.f;
#pragma unroll
  for (int c = 0; c < 4; ++c) { v[c] = __expf(v[c] - mx); sm += v[c]; }
  red[t] = sm; __syncthreads();
  for (int s = 64; s > 0; s >>= 1) { if (t < s) red[t] += red[t + s]; __syncthreads(); }
  float inv = 1.f / red[0];
#pragma unroll
  for (int c = 0; c < 4; ++c) L[t + c * 128] = v[c] * inv;
}

// ======================================================================
// K7: out[i, h*nch+col] = sum_j attn[h,i,j] * V[j, h*nch+col]   (per-head GEMM)
// ======================================================================
__global__ void k_attn_v(const float* __restrict__ attn, const float* __restrict__ V,
                         int ldv, int nch, float* __restrict__ out, int ldo) {
  const int lane = threadIdx.x;
  const int i0 = blockIdx.x * 16, nt = blockIdx.y, h = blockIdx.z;
  const int m = lane & 15, kh = lane >> 4, nn = lane & 15;
  const int col = nt * 16 + nn;
  v8f acc = {};
  for (int k0 = 0; k0 < NN; k0 += 32) {
    v16h a = load_frag_f32c(attn + (h * NN + i0 + m) * NN + k0, kh);
    v16h b;
#pragma unroll
    for (int e = 0; e < 16; ++e) {
      int k = k0 + kh * 16 + e;
      b[e] = (col < nch) ? (_Float16)V[k * ldv + h * nch + col] : (_Float16)0.f;
    }
    acc = wmma16(a, b, acc);
  }
  if (col < nch) {
#pragma unroll
    for (int r = 0; r < 8; ++r)
      out[(i0 + r + 8 * kh) * ldo + h * nch + col] = acc[r];
  }
}

// ======================================================================
// K8: res_pair: per i, (8x512 attn cols) @ (512x128 pair[i]) with WMMA
//     rows 8..15 of A are zero; streams pairwise exactly once (L2-resident)
// ======================================================================
__global__ void k_attn_pair(const float* __restrict__ attn, const float* __restrict__ pair,
                            float* __restrict__ results) {
  const int lane = threadIdx.x;
  const int i = blockIdx.x, nt = blockIdx.y;  // nt: 0..7 -> 16-col slab of 128
  const int m = lane & 15, kh = lane >> 4, nn = lane & 15;
  const float* Pi = pair + i * (NN * PDIM);
  v8f acc = {};
  for (int k0 = 0; k0 < NN; k0 += 32) {
    if (k0 + 32 < NN)
      __builtin_prefetch(Pi + (k0 + 32 + kh * 16) * PDIM + nt * 16 + nn, 0, 1);
    v16h a = {};
    if (m < NH)
      a = load_frag_f32c(attn + (m * NN + i) * NN + k0, kh);
    v16h b;
#pragma unroll
    for (int e = 0; e < 16; ++e)
      b[e] = (_Float16)Pi[(k0 + kh * 16 + e) * PDIM + nt * 16 + nn];
    acc = wmma16(a, b, acc);
  }
  if (kh == 0) {  // D rows 0..7 live in lanes 0..15; rows 8..15 are padding
#pragma unroll
    for (int r = 0; r < 8; ++r)
      results[i * RESW + NH * SV + NH * PV * 3 + NH * PV + r * PDIM + nt * 16 + nn] = acc[r];
  }
}

// ======================================================================
// K9: point output: local = R^T(g - t); write local coords + norms
// ======================================================================
__global__ void k_point_out(const float* __restrict__ respg, const float* __restrict__ rot,
                            const float* __restrict__ trans, float* __restrict__ results) {
  int t = blockIdx.x * blockDim.x + threadIdx.x;
  if (t >= NN * NH * PV) return;
  int i = t / (NH * PV);
  int hd = t % (NH * PV);
  int h = hd / PV, d = hd % PV;
  const float* g = respg + i * 192 + hd * 3;
  float x = g[0] - trans[i * 3 + 0];
  float y = g[1] - trans[i * 3 + 1];
  float z = g[2] - trans[i * 3 + 2];
  const float* R = rot + i * 9;  // out[r] = sum_c p[c] * R[r*3+c]
  float lx = R[0] * x + R[1] * y + R[2] * z;
  float ly = R[3] * x + R[4] * y + R[5] * z;
  float lz = R[6] * x + R[7] * y + R[8] * z;
  float* dst = results + i * RESW;
  dst[NH * SV + h * (PV * 3) + d * 3 + 0] = lx;
  dst[NH * SV + h * (PV * 3) + d * 3 + 1] = ly;
  dst[NH * SV + h * (PV * 3) + d * 3 + 2] = lz;
  dst[NH * SV + NH * PV * 3 + h * PV + d] = sqrtf(lx * lx + ly * ly + lz * lz + EPSN);
}

// ======================================================================
// K10: final GEMM: results (512 x 1408) @ Wout (1408 x 384) + bout
// ======================================================================
__global__ void k_final(const float* __restrict__ Rr, const float* __restrict__ Wout,
                        const float* __restrict__ bout, float* __restrict__ out) {
  const int lane = threadIdx.x;
  const int m0 = blockIdx.x * 16, n0 = blockIdx.y * 16;
  const int m = lane & 15, kh = lane >> 4, nn = lane & 15;
  v8f acc = {};
  for (int k0 = 0; k0 < RESW; k0 += 32) {
    v16h a = load_frag_f32c(Rr + (m0 + m) * RESW + k0, kh);
    v16h b;
#pragma unroll
    for (int e = 0; e < 16; ++e)
      b[e] = (_Float16)Wout[(k0 + kh * 16 + e) * DIM + n0 + nn];
    acc = wmma16(a, b, acc);
  }
  const float bb = bout[n0 + nn];
#pragma unroll
  for (int r = 0; r < 8; ++r)
    out[(m0 + r + 8 * kh) * DIM + n0 + nn] = acc[r] + bb;
}

// ======================================================================
extern "C" void kernel_launch(void* const* d_in, const int* in_sizes, int n_in,
                              void* d_out, int out_size, void* d_ws, size_t ws_size,
                              hipStream_t stream) {
  (void)in_sizes; (void)n_in; (void)out_size; (void)ws_size;
  const float* x     = (const float*)d_in[0];
  const float* pair  = (const float*)d_in[1];
  const float* rot   = (const float*)d_in[2];
  const float* trans = (const float*)d_in[3];
  const float* Wsq   = (const float*)d_in[4];
  const float* Wsk   = (const float*)d_in[5];
  const float* Wsv   = (const float*)d_in[6];
  const float* Wpq   = (const float*)d_in[7];
  const float* Wpk   = (const float*)d_in[8];
  const float* Wpv   = (const float*)d_in[9];
  const float* pw    = (const float*)d_in[10];
  const float* Wpair = (const float*)d_in[11];
  const float* bpair = (const float*)d_in[12];
  const float* Wout  = (const float*)d_in[13];
  const float* bout  = (const float*)d_in[14];
  float* out = (float*)d_out;

  float* wsf = (float*)d_ws;
  float* qs      = wsf + OFF_QS;
  float* ks      = wsf + OFF_KS;
  float* vs      = wsf + OFF_VS;
  float* qp      = wsf + OFF_QP;
  float* kp      = wsf + OFF_KP;
  float* vp      = wsf + OFF_VP;
  float* logits  = wsf + OFF_LOGITS;
  float* results = wsf + OFF_RESULTS;
  float* respg   = wsf + OFF_RESPG;
  float* rq      = wsf + OFF_RQ;
  float* ck      = wsf + OFF_CK;
  _Float16* Qt   = (_Float16*)(wsf + OFF_F32_END);
  _Float16* Kt   = Qt + HQT;

  // projections (WMMA GEMMs)
  k_gemm_xw<<<dim3(NN / 16,  8), 32, 0, stream>>>(x, Wsq, qs, DIM, 128);
  k_gemm_xw<<<dim3(NN / 16,  8), 32, 0, stream>>>(x, Wsk, ks, DIM, 128);
  k_gemm_xw<<<dim3(NN / 16,  8), 32, 0, stream>>>(x, Wsv, vs, DIM, 128);
  k_gemm_xw<<<dim3(NN / 16,  6), 32, 0, stream>>>(x, Wpq, qp, DIM, 96);
  k_gemm_xw<<<dim3(NN / 16,  6), 32, 0, stream>>>(x, Wpk, kp, DIM, 96);
  k_gemm_xw<<<dim3(NN / 16, 12), 32, 0, stream>>>(x, Wpv, vp, DIM, 192);

  // rotate points into global frame (in place)
  k_rotate<<<(NN * 32 + 255) / 256, 256, 0, stream>>>(qp, rot, trans, 32);
  k_rotate<<<(NN * 32 + 255) / 256, 256, 0, stream>>>(kp, rot, trans, 32);
  k_rotate<<<(NN * 64 + 255) / 256, 256, 0, stream>>>(vp, rot, trans, 64);

  // augmented q/k + distance biases
  k_prep<<<(NN * NH + 255) / 256, 256, 0, stream>>>(qs, ks, qp, kp, pw, Qt, Kt, rq, ck);

  // logits: pair bias first, then scalar+point accumulate, then softmax
  k_pairbias<<<dim3(NN * NN / 16), 32, 0, stream>>>(pair, Wpair, bpair, logits);
  k_qk_logits<<<dim3(NN / 16, NN / 16, NH), 32, 0, stream>>>(Qt, Kt, rq, ck, logits);
  k_softmax<<<dim3(NH * NN), 128, 0, stream>>>(logits);

  // outputs
  k_attn_v<<<dim3(NN / 16, 1, NH), 32, 0, stream>>>(logits, vs, 128, SV, results, RESW);
  k_attn_v<<<dim3(NN / 16, 2, NH), 32, 0, stream>>>(logits, vp, 192, PV * 3, respg, 192);
  k_attn_pair<<<dim3(NN, PDIM / 16), 32, 0, stream>>>(logits, pair, results);
  k_point_out<<<(NN * NH * PV + 255) / 256, 256, 0, stream>>>(respg, rot, trans, results);

  // final projection
  k_final<<<dim3(NN / 16, DIM / 16), 32, 0, stream>>>(results, Wout, bout, out);
}
